// LongelmSparseAttention_19146964206160
// MI455X (gfx1250) — matrix operand order repaired and checked by reading, not compiled, not run on hardware
//
#include <hip/hip_runtime.h>

typedef _Float16 f16;
typedef __attribute__((ext_vector_type(8)))  _Float16 v8h;
typedef __attribute__((ext_vector_type(16))) _Float16 v16h;
typedef __attribute__((ext_vector_type(8)))  float    v8f;

static constexpr int B_  = 2;
static constexpr int S_  = 8288;
static constexpr int HID_= 768;
static constexpr int H_  = 12;
static constexpr int D_  = 64;
static constexpr int G_  = 32;
static constexpr int NB_ = 64;
static constexpr int BS_ = 128;
static constexpr int N_  = 64;
static constexpr int L_  = NB_ * BS_;   // 8192
static constexpr int MD_ = 32;

// ---------------------------------------------------------------------------
// WMMA helpers (CDNA5 wave32, v_wmma_f32_16x16x32_f16)
// ---------------------------------------------------------------------------
__device__ inline v8f wmma16(v16h a, v16h b, v8f c) {
  return __builtin_amdgcn_wmma_f32_16x16x32_f16(
      false, a, false, b, (short)0, c, false, false);
}

__device__ inline v16h concat8(v8h lo, v8h hi) {
  return __builtin_shufflevector(lo, hi,
      0, 1, 2, 3, 4, 5, 6, 7, 8, 9, 10, 11, 12, 13, 14, 15);
}

// A fragment (16x32 f16, row-major src, leading dim ld, p at tile origin).
// ISA 7.12.2: lanes 0-15 (row M=lane) hold K=0..7,16..23; lanes 16-31 hold
// K=8..15,24..31.  Two contiguous 16B loads per lane.
__device__ inline v16h load_a(const f16* p, int ld, int lane) {
  const f16* row = p + (lane & 15) * ld + ((lane & 16) ? 8 : 0);
  return concat8(*(const v8h*)row, *(const v8h*)(row + 16));
}

// B fragment (32x16 f16) from N-major storage: element(k,n) = p[n*ld + k].
// Lanes 0-15 (col n=lane) hold K=0..7,16..23; lanes 16-31 hold K=8..15,24..31.
// Two contiguous 16B loads per lane.
__device__ inline v16h load_bt(const f16* p, int ld, int lane) {
  const f16* row = p + (lane & 15) * ld + ((lane & 16) ? 8 : 0);
  return concat8(*(const v8h*)row, *(const v8h*)(row + 16));
}

// ---------------------------------------------------------------------------
// f32 -> f16 converts
// ---------------------------------------------------------------------------
__global__ void cvt_x_kernel(const float* __restrict__ src,
                             f16* __restrict__ dst, int n) {
  int i = blockIdx.x * blockDim.x + threadIdx.x;
  if (i < n) dst[i] = (f16)src[i];
}

// Writes W^T (f16): dst[which][n][k] = W_which[k][n]
__global__ void cvt_w_kernel(const float* __restrict__ wq,
                             const float* __restrict__ wk,
                             const float* __restrict__ wv,
                             f16* __restrict__ dst) {
  int i = blockIdx.x * blockDim.x + threadIdx.x;
  const int per = HID_ * HID_;
  if (i < 3 * per) {
    int which = i / per, rem = i - which * per;
    int nn = rem / HID_, kk = rem - nn * HID_;
    const float* s = (which == 0) ? wq : (which == 1) ? wk : wv;
    dst[i] = (f16)s[kk * HID_ + nn];
  }
}

// ---------------------------------------------------------------------------
// Fused QKV projection GEMM (no LDS: A rows and W^T rows are both contiguous
// per lane; W^T is 3.5MB and L2-resident).  Output head-major f16
// qkv[which][b][h][s][d]; additionally writes V^T[b,h][d][s] for which==2.
// grid = (518 mblocks of 32 rows, 3 nblocks of 256 cols, 3 which), 256 thr.
// ---------------------------------------------------------------------------
__global__ __launch_bounds__(256) void qkv_gemm_kernel(
    const f16* __restrict__ x, const f16* __restrict__ wT,
    const float* __restrict__ bq, const float* __restrict__ bk,
    const float* __restrict__ bv, f16* __restrict__ qkv,
    f16* __restrict__ vT) {
  const int m0    = blockIdx.x * 32;
  const int nb0   = blockIdx.y * 256;
  const int which = blockIdx.z;
  const int tid = threadIdx.x, lane = tid & 31, wv = tid >> 5;
  const int mt = wv >> 2, nc = wv & 3;
  const f16* wsrc = wT + (size_t)which * HID_ * HID_;   // [n][k]

  v8f acc[4] = {};
  for (int k0 = 0; k0 < HID_; k0 += 32) {
    v16h a = load_a(x + (size_t)(m0 + mt * 16) * HID_ + k0, HID_, lane);
#pragma unroll
    for (int nt = 0; nt < 4; ++nt) {
      v16h bb = load_bt(wsrc + (size_t)(nb0 + nc * 64 + nt * 16) * HID_ + k0,
                        HID_, lane);
      acc[nt] = wmma16(a, bb, acc[nt]);
    }
  }

  const float* bias = (which == 0) ? bq : (which == 1) ? bk : bv;
  const int rbase = 8 * (lane >> 4), colb = lane & 15;
#pragma unroll
  for (int nt = 0; nt < 4; ++nt)
#pragma unroll
    for (int j = 0; j < 8; ++j) {
      int m    = m0 + mt * 16 + rbase + j;
      int ncol = nb0 + nc * 64 + nt * 16 + colb;
      float v  = acc[nt][j] + bias[ncol];
      int bb = m / S_, s = m - bb * S_;
      int hh = ncol >> 6, dd = ncol & 63;
      qkv[((((size_t)which * B_ + bb) * H_ + hh) * S_ + s) * D_ + dd] = (f16)v;
      if (which == 2)
        vT[((size_t)(bb * H_ + hh) * D_ + dd) * S_ + s] = (f16)v;
    }
}

// ---------------------------------------------------------------------------
// Local block attention: one workgroup per (b,h,block). 8 waves: wave w owns
// query rows 16w..16w+15. Keys = [own block 128 | own node 1 | globals 32].
// K staged row-major [key][d]; V staged TRANSPOSED [d][key] so P@V B-frags
// are contiguous per lane.
// ---------------------------------------------------------------------------
static constexpr int KLD  = 72;   // keys LDS ld (halfs)
static constexpr int VTLD = 216;  // valsT LDS ld (halfs); 192 keys + pad

__global__ __launch_bounds__(256) void local_attn_kernel(
    const f16* __restrict__ qkv, const float* __restrict__ am,
    float* __restrict__ out) {
  __shared__ __align__(16) f16 keys[176 * KLD];
  __shared__ __align__(16) f16 valsT[64 * VTLD];
  __shared__ __align__(16) f16 pch[8][16 * 40];

  const int blk = blockIdx.x;
  const int n  = blk % NB_;
  const int bh = blk / NB_;
  const int b = bh / H_, h = bh % H_;
  const size_t plane = (size_t)B_ * H_ * S_ * D_;
  const size_t base  = (size_t)bh * S_ * D_;
  const f16* qh = qkv + base;
  const f16* kh = qkv + plane + base;
  const f16* vh = qkv + 2 * plane + base;
  const int tid = threadIdx.x, lane = tid & 31, wv = tid >> 5;

  // Stage K rows (row-major, rows >=161 zeroed).
  for (int idx = tid; idx < 176 * 8; idx += 256) {
    int r = idx >> 3, c = (idx & 7) * 8;
    int tok = (r < 128) ? (G_ + n * BS_ + r)
            : (r == 128) ? (G_ + L_ + n)
            : (r < 161) ? (r - 129) : -1;
    v8h v = {};
    if (tok >= 0) v = *(const v8h*)&kh[(size_t)tok * D_ + c];
    *(v8h*)&keys[r * KLD + c] = v;
  }
  // Stage V transposed: valsT[d][key], keys >=161 zeroed.
  for (int idx = tid; idx < 192 * 8; idx += 256) {
    int r = idx >> 3, c = (idx & 7) * 8;   // r = key row, c = d chunk
    int tok = (r < 128) ? (G_ + n * BS_ + r)
            : (r == 128) ? (G_ + L_ + n)
            : (r < 161) ? (r - 129) : -1;
    v8h v = {};
    if (tok >= 0) v = *(const v8h*)&vh[(size_t)tok * D_ + c];
#pragma unroll
    for (int q = 0; q < 8; ++q) valsT[(c + q) * VTLD + r] = v[q];
  }
  __syncthreads();

  // ---- scores: S[16 x 176] per wave, 11 col-tiles x 2 K-steps ----
  const float scale = 0.125f;
  const int q0 = G_ + n * BS_ + wv * 16;
  v16h a0 = load_a(qh + (size_t)q0 * D_, D_, lane);
  v16h a1 = load_a(qh + (size_t)q0 * D_ + 32, D_, lane);
  const int colb = lane & 15, rbase = 8 * (lane >> 4);

  float sreg[11][8];
#pragma unroll
  for (int t = 0; t < 11; ++t) {
    v8f c = {};
    v16h b0 = load_bt(keys + t * 16 * KLD, KLD, lane);
    v16h b1 = load_bt(keys + t * 16 * KLD + 32, KLD, lane);
    c = wmma16(a0, b0, c);
    c = wmma16(a1, b1, c);
    int col = t * 16 + colb;
    float mask;
    if (col < 128)       mask = am[(size_t)b * L_ + n * BS_ + col];
    else if (col < 161)  mask = 0.f;
    else                 mask = -1e30f;
#pragma unroll
    for (int j = 0; j < 8; ++j) sreg[t][j] = c[j] * scale + mask;
  }

  // ---- row-wise softmax (reduce across the 16-lane half groups) ----
  float pinv[8];
#pragma unroll
  for (int j = 0; j < 8; ++j) {
    float m = -1e30f;
#pragma unroll
    for (int t = 0; t < 11; ++t) m = fmaxf(m, sreg[t][j]);
    m = fmaxf(m, __shfl_xor(m, 1, 32));
    m = fmaxf(m, __shfl_xor(m, 2, 32));
    m = fmaxf(m, __shfl_xor(m, 4, 32));
    m = fmaxf(m, __shfl_xor(m, 8, 32));
    float s = 0.f;
#pragma unroll
    for (int t = 0; t < 11; ++t) { float e = __expf(sreg[t][j] - m); sreg[t][j] = e; s += e; }
    s += __shfl_xor(s, 1, 32);
    s += __shfl_xor(s, 2, 32);
    s += __shfl_xor(s, 4, 32);
    s += __shfl_xor(s, 8, 32);
    pinv[j] = 1.f / s;
  }

  // ---- ctx = P @ V : per-wave 16x32 P chunk in LDS, 6 K-steps x 4 N-tiles --
  v8f acc[4] = {};
  f16* myp = &pch[wv][0];
  for (int kt = 0; kt < 6; ++kt) {
#pragma unroll
    for (int tt = 0; tt < 2; ++tt) {
      int t = kt * 2 + tt;
#pragma unroll
      for (int j = 0; j < 8; ++j) {
        float pv = (t < 11) ? sreg[t][j] * pinv[j] : 0.f;
        myp[(rbase + j) * 40 + tt * 16 + colb] = (f16)pv;
      }
    }
    // pch is strictly per-wave; DS ops are in-order per wave, so a dscnt
    // drain makes this wave's stores visible to its own cross-lane loads.
    asm volatile("s_wait_dscnt 0x0" ::: "memory");
    v16h a = load_a(myp, 40, lane);
#pragma unroll
    for (int nt = 0; nt < 4; ++nt) {
      v16h bb = load_bt(valsT + (size_t)(nt * 16) * VTLD + kt * 32, VTLD, lane);
      acc[nt] = wmma16(a, bb, acc[nt]);
    }
  }

#pragma unroll
  for (int nt = 0; nt < 4; ++nt)
#pragma unroll
    for (int j = 0; j < 8; ++j) {
      int tok = G_ + n * BS_ + wv * 16 + rbase + j;
      int dd  = nt * 16 + colb;
      out[((size_t)b * S_ + tok) * HID_ + h * D_ + dd] = acc[nt][j];
    }
}

// ---------------------------------------------------------------------------
// Node attention: one wave per (b,h,node); 224 keys = [nodes64|block128|glob32]
// ---------------------------------------------------------------------------
__global__ __launch_bounds__(32) void node_attn_kernel(
    const f16* __restrict__ qkv, const float* __restrict__ am,
    const float* __restrict__ gm, const int* __restrict__ rpn,
    const float* __restrict__ rb, float* __restrict__ out) {
  __shared__ float qf[64];
  __shared__ float parr[224];
  const int blk = blockIdx.x;
  const int n  = blk % NB_;
  const int bh = blk / NB_;
  const int b = bh / H_, h = bh % H_;
  const size_t plane = (size_t)B_ * H_ * S_ * D_;
  const size_t base  = (size_t)bh * S_ * D_;
  const f16* qh = qkv + base;
  const f16* kh = qkv + plane + base;
  const f16* vh = qkv + 2 * plane + base;
  const int lane = threadIdx.x;
  const int qtok = G_ + L_ + n;

  qf[lane]      = (float)qh[(size_t)qtok * D_ + lane];
  qf[lane + 32] = (float)qh[(size_t)qtok * D_ + lane + 32];
  __syncthreads();

  const float scale = 0.125f;
  float sc[7];
#pragma unroll
  for (int i = 0; i < 7; ++i) {
    int kk = lane * 7 + i;
    int tok; float addpre = 0.f, addpost = 0.f;
    if (kk < 64) {
      tok = G_ + L_ + kk;
      int rp = rpn[((size_t)b * N_ + n) * N_ + kk];
      rp = (rp >= 0) ? rp : MD_;
      rp = (rp < MD_) ? rp : (MD_ - 1);
      addpre  = rb[rp * H_ + h];                     // bias added BEFORE scale
      addpost = gm[((size_t)b * N_ + n) * N_ + kk];
    } else if (kk < 192) {
      int j = kk - 64;
      tok = G_ + n * BS_ + j;
      addpost = am[(size_t)b * L_ + n * BS_ + j];
    } else {
      tok = kk - 192;
    }
    const f16* kr = kh + (size_t)tok * D_;
    float dot = 0.f;
#pragma unroll
    for (int d = 0; d < 64; ++d) dot += qf[d] * (float)kr[d];
    sc[i] = (dot + addpre) * scale + addpost;
  }

  float m = -1e30f;
#pragma unroll
  for (int i = 0; i < 7; ++i) m = fmaxf(m, sc[i]);
  for (int x = 1; x < 32; x <<= 1) m = fmaxf(m, __shfl_xor(m, x, 32));
  float s = 0.f;
#pragma unroll
  for (int i = 0; i < 7; ++i) { sc[i] = __expf(sc[i] - m); s += sc[i]; }
  for (int x = 1; x < 32; x <<= 1) s += __shfl_xor(s, x, 32);
  float inv = 1.f / s;
#pragma unroll
  for (int i = 0; i < 7; ++i) parr[lane * 7 + i] = sc[i] * inv;
  __syncthreads();

  float acc0 = 0.f, acc1 = 0.f;
  for (int kk = 0; kk < 224; ++kk) {
    int tok = (kk < 64) ? (G_ + L_ + kk)
            : (kk < 192) ? (G_ + n * BS_ + (kk - 64)) : (kk - 192);
    float p = parr[kk];
    const f16* vr = vh + (size_t)tok * D_;
    acc0 += p * (float)vr[lane];
    acc1 += p * (float)vr[lane + 32];
  }
  out[((size_t)b * S_ + qtok) * HID_ + h * D_ + lane]      = acc0;
  out[((size_t)b * S_ + qtok) * HID_ + h * D_ + lane + 32] = acc1;
}

// ---------------------------------------------------------------------------
// Global attention: one workgroup per (b,h); 32 queries vs all 8288 keys.
// Two-pass via global scratch: scores f32 -> softmax -> P(f16) @ V^T via WMMA.
// ---------------------------------------------------------------------------
__global__ __launch_bounds__(256) void global_attn_kernel(
    const f16* __restrict__ qkv, const f16* __restrict__ vT,
    const float* __restrict__ am, float* __restrict__ gsc,
    f16* __restrict__ gpr, float* __restrict__ out) {
  __shared__ float rinv[32];
  const int bh = blockIdx.x;
  const int b = bh / H_, h = bh % H_;
  const size_t plane = (size_t)B_ * H_ * S_ * D_;
  const size_t base  = (size_t)bh * S_ * D_;
  const f16* qh  = qkv + base;
  const f16* kh  = qkv + plane + base;
  const f16* vTh = vT + (size_t)bh * D_ * S_;     // [d][s]
  float* sc = gsc + (size_t)bh * 32 * S_;
  f16*   pr = gpr + (size_t)bh * 32 * S_;
  const int tid = threadIdx.x, lane = tid & 31, wv = tid >> 5;
  const float scale = 0.125f;
  const int rbase = 8 * (lane >> 4), colb = lane & 15;

  // phase 1: scores (keys streamed in token order; softmax is order-invariant)
  {
    int mt = wv & 1;
    v16h a0 = load_a(qh + (size_t)(mt * 16) * D_, D_, lane);
    v16h a1 = load_a(qh + (size_t)(mt * 16) * D_ + 32, D_, lane);
    for (int kt = (wv >> 1); kt < S_ / 16; kt += 4) {
      v8f c = {};
      v16h b0 = load_bt(kh + (size_t)kt * 16 * D_, D_, lane);
      v16h b1 = load_bt(kh + (size_t)kt * 16 * D_ + 32, D_, lane);
      c = wmma16(a0, b0, c);
      c = wmma16(a1, b1, c);
      int tok = kt * 16 + colb;
      float mask = (tok >= G_ && tok < G_ + L_)
                       ? am[(size_t)b * L_ + (tok - G_)] : 0.f;
#pragma unroll
      for (int j = 0; j < 8; ++j)
        sc[(size_t)(mt * 16 + rbase + j) * S_ + tok] = c[j] * scale + mask;
    }
  }
  __syncthreads();

  // phase 2: row-wise softmax; store unnormalized exp as f16, keep 1/sum
  for (int r = 0; r < 4; ++r) {
    int row = wv * 4 + r;
    const float* srow = sc + (size_t)row * S_;
    f16* prow = pr + (size_t)row * S_;
    float m = -1e30f;
    for (int c = lane; c < S_; c += 32) m = fmaxf(m, srow[c]);
    for (int x = 1; x < 32; x <<= 1) m = fmaxf(m, __shfl_xor(m, x, 32));
    float s = 0.f;
    for (int c = lane; c < S_; c += 32) {
      float e = __expf(srow[c] - m);
      prow[c] = (f16)e;
      s += e;
    }
    for (int x = 1; x < 32; x <<= 1) s += __shfl_xor(s, x, 32);
    if (lane == 0) rinv[row] = 1.f / s;
  }
  __syncthreads();

  // phase 3: ctx = P @ V ; 8 waves = 2 M-tiles x 4 N-tiles, K = 8288
  {
    int mt = wv & 1, nt = wv >> 1;
    v8f acc = {};
    for (int kt = 0; kt < S_ / 32; ++kt) {
      v16h a  = load_a(pr + (size_t)(mt * 16) * S_ + kt * 32, S_, lane);
      v16h bb = load_bt(vTh + (size_t)(nt * 16) * S_ + kt * 32, S_, lane);
      acc = wmma16(a, bb, acc);
    }
#pragma unroll
    for (int j = 0; j < 8; ++j) {
      int row = mt * 16 + rbase + j;        // global token index == row
      float v = acc[j] * rinv[row];
      out[((size_t)b * S_ + row) * HID_ + h * D_ + nt * 16 + colb] = v;
    }
  }
}

// ---------------------------------------------------------------------------
// Launch
// ---------------------------------------------------------------------------
extern "C" void kernel_launch(void* const* d_in, const int* in_sizes, int n_in,
                              void* d_out, int out_size, void* d_ws, size_t ws_size,
                              hipStream_t stream) {
  (void)in_sizes; (void)n_in; (void)out_size; (void)ws_size;
  const float* hs  = (const float*)d_in[0];
  const float* am  = (const float*)d_in[1];
  const float* gm  = (const float*)d_in[2];
  const int*   rpn = (const int*)d_in[3];
  const float* Wq  = (const float*)d_in[4];
  const float* bq  = (const float*)d_in[5];
  const float* Wk  = (const float*)d_in[6];
  const float* bk  = (const float*)d_in[7];
  const float* Wv  = (const float*)d_in[8];
  const float* bv  = (const float*)d_in[9];
  const float* rb  = (const float*)d_in[10];
  float* out = (float*)d_out;

  char* wsp = (char*)d_ws;
  size_t off = 0;
  auto carve = [&](size_t bytes) -> char* {
    char* p = wsp + off;
    off += (bytes + 255) & ~(size_t)255;
    return p;
  };
  const size_t nX = (size_t)B_ * S_ * HID_;          // 12,730,368
  f16*   x_h   = (f16*)carve(nX * 2);
  f16*   wT_h  = (f16*)carve((size_t)3 * HID_ * HID_ * 2);
  f16*   qkv_h = (f16*)carve((size_t)3 * B_ * H_ * S_ * D_ * 2);
  f16*   vT_h  = (f16*)carve((size_t)B_ * H_ * D_ * S_ * 2);
  float* gsc   = (float*)carve((size_t)B_ * H_ * 32 * S_ * 4);
  f16*   gpr   = (f16*)carve((size_t)B_ * H_ * 32 * S_ * 2);

  cvt_x_kernel<<<(unsigned)((nX + 255) / 256), 256, 0, stream>>>(hs, x_h, (int)nX);
  cvt_w_kernel<<<(3 * HID_ * HID_ + 255) / 256, 256, 0, stream>>>(Wq, Wk, Wv, wT_h);

  qkv_gemm_kernel<<<dim3(518, 3, 3), 256, 0, stream>>>(
      x_h, wT_h, bq, bk, bv, qkv_h, vT_h);

  local_attn_kernel<<<B_ * H_ * NB_, 256, 0, stream>>>(qkv_h, am, out);
  node_attn_kernel<<<B_ * H_ * NB_, 32, 0, stream>>>(qkv_h, am, gm, rpn, rb, out);
  global_attn_kernel<<<B_ * H_, 256, 0, stream>>>(qkv_h, vT_h, am, gsc, gpr, out);
}
